// DCGRUCell_66133906423964
// MI455X (gfx1250) — matrix operand order repaired and checked by reading, not compiled, not run on hardware
//
#include <hip/hip_runtime.h>
#include <hip/hip_bf16.h>
#include <math.h>

// Problem constants (from reference): B=64, N=2048, IN=2, OUT=64, J=4
#define B_   64
#define N_   2048
#define J_   4
#define FIN_ 66      // IN + OUT
#define O_   64

typedef float  v8f   __attribute__((ext_vector_type(8)));
typedef __bf16 v8bf  __attribute__((ext_vector_type(8)));
typedef __bf16 v16bf __attribute__((ext_vector_type(16)));

// Fast gate nonlinearities: single v_rcp_f32 instead of the IEEE div chain.
__device__ __forceinline__ float sigmoid_fast(float x) {
    return __builtin_amdgcn_rcpf(1.0f + __expf(-x));
}
__device__ __forceinline__ float tanh_fast(float x) {
    // tanh(x) = 1 - 2/(exp(2x)+1)
    return 1.0f - 2.0f * __builtin_amdgcn_rcpf(__expf(2.0f * x) + 1.0f);
}

// ---------------------------------------------------------------------------
// Kernel 0: f32 -> bf16 conversion of the supports matrices (streamed once).
// ---------------------------------------------------------------------------
__global__ void __launch_bounds__(256)
cvt_f32_bf16_kernel(const float* __restrict__ in, __bf16* __restrict__ out, long n) {
    long i = (long)blockIdx.x * blockDim.x + threadIdx.x;
    if (i < n) out[i] = (__bf16)in[i];
}

// ---------------------------------------------------------------------------
// Kernel 1/3: Yt[b][col][j*N + n] = sum_l xs[b][n][l] * W[j][l][col_in_gate]
// xs = [input(2) | sbuf(64)];  W layout [J][66][64].
// Stored K-major so the WMMA B fragment reads contiguous K runs.
// Consecutive threads share (b,col,j) -> W loads are wave-uniform (SGPR).
// ---------------------------------------------------------------------------
__global__ void __launch_bounds__(256)
yt_kernel(const float* __restrict__ inp, const float* __restrict__ sbuf,
          const float* __restrict__ W0,  const float* __restrict__ W1,
          __bf16* __restrict__ Yt, int ncols, long total) {
    long idx = (long)blockIdx.x * blockDim.x + threadIdx.x;
    if (idx >= total) return;
    int  n   = (int)(idx % N_);  long t = idx / N_;
    int  j   = (int)(t % J_);    t /= J_;
    int  col = (int)(t % ncols);
    int  b   = (int)(t / ncols);

    const float* W  = (col < O_) ? W0 : W1;
    const int    c  = (col < O_) ? col : col - O_;
    const float* Wj = W + (long)j * FIN_ * O_;
    const float* xi = inp  + ((long)b * N_ + n) * 2;
    const float* xs = sbuf + ((long)b * N_ + n) * O_;

    float acc = xi[0] * Wj[c] + xi[1] * Wj[O_ + c];
#pragma unroll 8
    for (int l = 0; l < O_; ++l)
        acc += xs[l] * Wj[(2 + l) * O_ + c];

    Yt[((long)b * ncols + col) * (J_ * N_) + (long)j * N_ + n] = (__bf16)acc;
}

// ---------------------------------------------------------------------------
// WMMA fragment helpers (wave32, v_wmma_f32_16x16x32_bf16).
// A (16x32 bf16): lane L<16 holds row M=L, K in {ka..ka+7} U {ka+16..ka+23},
//                 lanes 16-31 hold rows 0-15 with ka += 8 (ISA 7.12.2).
// B (32x16 bf16): lane L holds column N = L&15; same K split as A:
//                 K in {kb..kb+7} U {kb+16..kb+23}, kb += 8 for lanes>=16.
// ---------------------------------------------------------------------------
__device__ __forceinline__ v16bf load_a_frag(const __bf16* Arow, int m, int hi) {
    const int ka = m + hi * 8;
    v8bf lo = *(const v8bf*)(Arow + ka);
    v8bf hh = *(const v8bf*)(Arow + ka + 16);
    v16bf a;
#pragma unroll
    for (int t = 0; t < 8; ++t) { a[t] = lo[t]; a[t + 8] = hh[t]; }
    return a;
}

__device__ __forceinline__ v16bf load_b_frag(const __bf16* Bcol, int m, int hi) {
    const __bf16* p = Bcol + m + hi * 8;
    v8bf lo = *(const v8bf*)(p);
    v8bf hh = *(const v8bf*)(p + 16);
    v16bf bfrag;
#pragma unroll
    for (int t = 0; t < 8; ++t) { bfrag[t] = lo[t]; bfrag[t + 8] = hh[t]; }
    return bfrag;
}

#define WMMA_BF16(A, Bf, C) \
    __builtin_amdgcn_wmma_f32_16x16x32_bf16(false, (A), false, (Bf), (short)0, (C), false, false)

// ---------------------------------------------------------------------------
// Kernel 2: fused r+u gates.
// Block = 8 waves covering a 64-row x 128-col output tile.
// Wave (rg, cg) = (wave>>2, wave&3) owns a 32x32 sub-tile:
//   4 accumulators (2 row frags x 2 col frags), per K-step 2 A + 2 B frag
//   loads feed 4 v_wmma -> 1 KB loaded per WMMA (2x better density).
// cols 0-63 = r gate, 64-127 = u gate (cg>=2).
// ---------------------------------------------------------------------------
__global__ void __launch_bounds__(256)
gc_ru_kernel(const __bf16* __restrict__ S, const __bf16* __restrict__ Yt,
             const float* __restrict__ state,
             const float* __restrict__ br, const float* __restrict__ bu,
             float* __restrict__ rstate, float* __restrict__ uout) {
    const int lane = threadIdx.x & 31;
    const int wave = threadIdx.x >> 5;
    const int rg   = wave >> 2;          // 0..1 row group (32 rows)
    const int cg   = wave & 3;           // 0..3 col group (32 cols)
    const int rt   = blockIdx.x;         // 64-row tile (N/64)
    const int b    = blockIdx.y;         // batch
    const int lrow = lane & 15;
    const int hi   = lane >> 4;

    const long KW   = (long)J_ * N_;
    const int  row0 = rt * 64 + rg * 32;
    const int  col0 = cg * 32;

    const __bf16* A0 = S  + (long)(row0 + lrow) * N_;
    const __bf16* A1 = A0 + 16L * N_;
    const __bf16* B0 = Yt + ((long)b * 128 + col0 + lrow) * KW;
    const __bf16* B1 = B0 + 16L * KW;

    v8f acc00 = {}, acc01 = {}, acc10 = {}, acc11 = {};
#pragma unroll 1
    for (int j = 0; j < J_; ++j) {
        const long  aoff = (long)j * N_ * N_;
        const long  boff = (long)j * N_;
#pragma unroll 2
        for (int m = 0; m < N_; m += 32) {
            __builtin_prefetch(A0 + aoff + m + 256, 0, 1);   // global_prefetch_b8
            v16bf a0 = load_a_frag(A0 + aoff, m, hi);
            v16bf a1 = load_a_frag(A1 + aoff, m, hi);
            v16bf b0 = load_b_frag(B0 + boff, m, hi);
            v16bf b1 = load_b_frag(B1 + boff, m, hi);
            acc00 = WMMA_BF16(a0, b0, acc00);
            acc01 = WMMA_BF16(a0, b1, acc01);
            acc10 = WMMA_BF16(a1, b0, acc10);
            acc11 = WMMA_BF16(a1, b1, acc11);
        }
    }

    const bool  isU   = cg >= 2;
    const int   gcol0 = (cg & 1) * 32 + lrow;       // column within the gate
    const int   gcol1 = gcol0 + 16;
    const float bias0 = isU ? bu[gcol0] : br[gcol0];
    const float bias1 = isU ? bu[gcol1] : br[gcol1];

#pragma unroll
    for (int v = 0; v < 8; ++v) {
        const int  r0   = row0 + v + hi * 8;
        const int  r1   = r0 + 16;
        const long i00  = ((long)b * N_ + r0) * O_ + gcol0;
        const long i01  = ((long)b * N_ + r0) * O_ + gcol1;
        const long i10  = ((long)b * N_ + r1) * O_ + gcol0;
        const long i11  = ((long)b * N_ + r1) * O_ + gcol1;
        const float g00 = sigmoid_fast(acc00[v] + bias0);
        const float g01 = sigmoid_fast(acc01[v] + bias1);
        const float g10 = sigmoid_fast(acc10[v] + bias0);
        const float g11 = sigmoid_fast(acc11[v] + bias1);
        if (isU) {
            uout[i00] = g00; uout[i01] = g01; uout[i10] = g10; uout[i11] = g11;
        } else {
            rstate[i00] = g00 * state[i00];
            rstate[i01] = g01 * state[i01];
            rstate[i10] = g10 * state[i10];
            rstate[i11] = g11 * state[i11];
        }
    }
}

// ---------------------------------------------------------------------------
// Kernel 4: c gate + GRU combine.
// Block = 4 waves covering a 64-row x 64-col output tile; wave (rg,cgr) =
// (wave>>1, wave&1) owns a 32x32 sub-tile with 4 accumulators.
// ---------------------------------------------------------------------------
__global__ void __launch_bounds__(128)
gc_c_kernel(const __bf16* __restrict__ S, const __bf16* __restrict__ Yt,
            const float* __restrict__ state, const float* __restrict__ bc,
            const float* __restrict__ uin, float* __restrict__ out) {
    const int lane = threadIdx.x & 31;
    const int wave = threadIdx.x >> 5;
    const int rg   = wave >> 1;          // 0..1
    const int cgr  = wave & 1;           // 0..1
    const int rt   = blockIdx.x;
    const int b    = blockIdx.y;
    const int lrow = lane & 15;
    const int hi   = lane >> 4;

    const long KW   = (long)J_ * N_;
    const int  row0 = rt * 64 + rg * 32;
    const int  col0 = cgr * 32;

    const __bf16* A0 = S  + (long)(row0 + lrow) * N_;
    const __bf16* A1 = A0 + 16L * N_;
    const __bf16* B0 = Yt + ((long)b * O_ + col0 + lrow) * KW;
    const __bf16* B1 = B0 + 16L * KW;

    v8f acc00 = {}, acc01 = {}, acc10 = {}, acc11 = {};
#pragma unroll 1
    for (int j = 0; j < J_; ++j) {
        const long aoff = (long)j * N_ * N_;
        const long boff = (long)j * N_;
#pragma unroll 2
        for (int m = 0; m < N_; m += 32) {
            __builtin_prefetch(A0 + aoff + m + 256, 0, 1);
            v16bf a0 = load_a_frag(A0 + aoff, m, hi);
            v16bf a1 = load_a_frag(A1 + aoff, m, hi);
            v16bf b0 = load_b_frag(B0 + boff, m, hi);
            v16bf b1 = load_b_frag(B1 + boff, m, hi);
            acc00 = WMMA_BF16(a0, b0, acc00);
            acc01 = WMMA_BF16(a0, b1, acc01);
            acc10 = WMMA_BF16(a1, b0, acc10);
            acc11 = WMMA_BF16(a1, b1, acc11);
        }
    }

    const int   gcol0 = col0 + lrow;
    const int   gcol1 = gcol0 + 16;
    const float bias0 = bc[gcol0];
    const float bias1 = bc[gcol1];

#pragma unroll
    for (int v = 0; v < 8; ++v) {
        const int  r0  = row0 + v + hi * 8;
        const int  r1  = r0 + 16;
        const long i00 = ((long)b * N_ + r0) * O_ + gcol0;
        const long i01 = ((long)b * N_ + r0) * O_ + gcol1;
        const long i10 = ((long)b * N_ + r1) * O_ + gcol0;
        const long i11 = ((long)b * N_ + r1) * O_ + gcol1;
        const float c00 = tanh_fast(acc00[v] + bias0);
        const float c01 = tanh_fast(acc01[v] + bias1);
        const float c10 = tanh_fast(acc10[v] + bias0);
        const float c11 = tanh_fast(acc11[v] + bias1);
        float u;
        u = uin[i00]; out[i00] = u * state[i00] + (1.0f - u) * c00;
        u = uin[i01]; out[i01] = u * state[i01] + (1.0f - u) * c01;
        u = uin[i10]; out[i10] = u * state[i10] + (1.0f - u) * c10;
        u = uin[i11]; out[i11] = u * state[i11] + (1.0f - u) * c11;
    }
}

// ---------------------------------------------------------------------------
// Host-side orchestration.
// Workspace layout (bytes):
//   [0,    32MB)  supports bf16           J*N*N*2      = 33554432
//   [32MB, 160MB) Yt (K-major bf16)       B*128*J*N*2  = 134217728  (reused for c)
//   [160MB,192MB) u  (f32)                B*N*64*4     = 33554432
//   [192MB,224MB) r*state (f32)           B*N*64*4     = 33554432
// ---------------------------------------------------------------------------
extern "C" void kernel_launch(void* const* d_in, const int* in_sizes, int n_in,
                              void* d_out, int out_size, void* d_ws, size_t ws_size,
                              hipStream_t stream) {
    const float* input    = (const float*)d_in[0];
    const float* state    = (const float*)d_in[1];
    const float* supports = (const float*)d_in[2];
    const float* Wr       = (const float*)d_in[3];
    const float* br       = (const float*)d_in[4];
    const float* Wu       = (const float*)d_in[5];
    const float* bu       = (const float*)d_in[6];
    const float* Wc       = (const float*)d_in[7];
    const float* bc       = (const float*)d_in[8];
    float*       out      = (float*)d_out;

    char* ws = (char*)d_ws;
    __bf16* Sb     = (__bf16*)(ws);
    __bf16* Yt     = (__bf16*)(ws + 33554432L);
    float*  uo     = (float*) (ws + 33554432L + 134217728L);
    float*  rstate = (float*) (ws + 33554432L + 134217728L + 33554432L);

    // 0) supports -> bf16
    {
        long n = (long)J_ * N_ * N_;
        cvt_f32_bf16_kernel<<<(unsigned)((n + 255) / 256), 256, 0, stream>>>(supports, Sb, n);
    }
    // 1) Yt for r,u (128 fused columns)
    {
        long total = (long)B_ * 128 * J_ * N_;
        yt_kernel<<<(unsigned)((total + 255) / 256), 256, 0, stream>>>(
            input, state, Wr, Wu, Yt, 128, total);
    }
    // 2) big WMMA: r,u gates -> r*state, u
    gc_ru_kernel<<<dim3(N_ / 64, B_), 256, 0, stream>>>(Sb, Yt, state, br, bu, rstate, uo);
    // 3) Yt for c (64 columns), from [input | r*state]
    {
        long total = (long)B_ * O_ * J_ * N_;
        yt_kernel<<<(unsigned)((total + 255) / 256), 256, 0, stream>>>(
            input, rstate, Wc, Wc, Yt, O_, total);
    }
    // 4) big WMMA: c gate + GRU combine -> out
    gc_c_kernel<<<dim3(N_ / 64, B_), 128, 0, stream>>>(Sb, Yt, state, bc, uo, out);
}
